// MGDN_71073118814872
// MI455X (gfx1250) — matrix operand order repaired and checked by reading, not compiled, not run on hardware
//
#include <hip/hip_runtime.h>
#include <hip/hip_bf16.h>

// ---------------------------------------------------------------------------
// MGDN forward for MI455X (gfx1250, wave32, WMMA).
// Exploits: constant in-degree (21) => GCN norm == 1/21 everywhere.
// GEMMs run on v_wmma_f32_16x16x32_f16 (f16 in, f32 accum), tiles via LDS.
// Fragment loads are vectorized (ds_load_b128) via K-contiguous LDS layouts.
// ---------------------------------------------------------------------------

typedef _Float16 half16 __attribute__((ext_vector_type(16)));
typedef _Float16 half8  __attribute__((ext_vector_type(8)));
typedef float    float8 __attribute__((ext_vector_type(8)));

#define BB   128      // batch
#define NN   1000     // nodes per graph
#define MM   (BB*NN)  // 128000 rows
#define TOPK 20
#define DIM  64

__device__ __forceinline__ float bn_inv() { return rsqrtf(1.0f + 1e-5f); }

// ---------------------------------------------------------------------------
// Top-20 cosine neighbors per row of emb[N][64]. One block per row.
// Ranking by dot/|w_j| (dividing by |w_i| does not change per-row order).
// Tie-break: lowest index (matches lax.top_k).
// ---------------------------------------------------------------------------
__global__ __launch_bounds__(256)
void topk_kernel(const float* __restrict__ emb, int* __restrict__ idxOut) {
  __shared__ float wi[DIM];
  __shared__ float cosv[NN];
  __shared__ float rv[256];
  __shared__ int   ri[256];
  const int i = blockIdx.x;
  const int tid = threadIdx.x;
  if (tid < DIM) wi[tid] = emb[i * DIM + tid];
  __syncthreads();
  for (int j = tid; j < NN; j += 256) {
    const float* wj = emb + j * DIM;
    float dot = 0.f, nj = 0.f;
#pragma unroll
    for (int d = 0; d < DIM; ++d) { float v = wj[d]; dot += wi[d] * v; nj += v * v; }
    cosv[j] = dot * rsqrtf(nj);
  }
  __syncthreads();
  for (int t = 0; t < TOPK; ++t) {
    float bv = -3.0e38f; int bi = 0x7fffffff;
    for (int j = tid; j < NN; j += 256)
      if (cosv[j] > bv) { bv = cosv[j]; bi = j; }   // strided ascending => ties keep low idx
    rv[tid] = bv; ri[tid] = bi;
    __syncthreads();
    for (int off = 128; off > 0; off >>= 1) {
      if (tid < off) {
        if (rv[tid + off] > rv[tid] ||
            (rv[tid + off] == rv[tid] && ri[tid + off] < ri[tid])) {
          rv[tid] = rv[tid + off]; ri[tid] = ri[tid + off];
        }
      }
      __syncthreads();
    }
    if (tid == 0) { idxOut[i * TOPK + t] = ri[0]; cosv[ri[0]] = -3.4e38f; }
    __syncthreads();
  }
}

// ---------------------------------------------------------------------------
// WMMA GEMM: Out[M][NC] = A[M][lda](cols aoff..aoff+K) @ Wt[K][NC] (+bias)
// 256 threads = 8 waves; each wave owns one 16x16 output tile.
// A staged row-major [ROWS][KPAD] f16; W staged TRANSPOSED [NC][KPAD] f16 so
// both fragments are K-contiguous per lane => half8 loads -> ds_load_b128.
// Requires M % ROWS == 0 (true for all call sites: 128000 % {32,128} == 0).
// Fragment layouts per CDNA5 ISA 7.12.2 (16-bit A 16x32, B 32x16, f32 C/D).
// ---------------------------------------------------------------------------
template <int KPAD, int NC, int ROWS, bool HAS_BIAS>
__global__ __launch_bounds__(256)
void wmma_gemm(const float* __restrict__ Aall, int lda, int aoff, int K,
               const float* __restrict__ Wt, const float* __restrict__ bias,
               float* __restrict__ Out) {
  __shared__ _Float16 sA[ROWS * KPAD];     // [row][k]
  __shared__ _Float16 sWt[NC * KPAD];      // [col][k]  (transposed W)
  const int tid = threadIdx.x;
  const int row0 = blockIdx.x * ROWS;

  for (int u = tid; u < ROWS * KPAD; u += 256) {
    int r = u / KPAD, k = u - r * KPAD;
    float v = (k < K) ? Aall[(size_t)(row0 + r) * lda + aoff + k] : 0.0f;
    sA[u] = (_Float16)v;
  }
  for (int u = tid; u < NC * KPAD; u += 256) {
    int n = u / KPAD, k = u - n * KPAD;
    sWt[u] = (_Float16)((k < K) ? Wt[k * NC + n] : 0.0f);
  }
  __syncthreads();

  const int wave = tid >> 5;
  const int lane = tid & 31;
  const int half = lane >> 4;
  const int rc   = lane & 15;
  constexpr int CT = NC / 16;        // column tiles
  const int rt = wave / CT;          // row tile of this wave
  const int ct = wave % CT;

  const _Float16* aBase = &sA[(rt * 16 + rc) * KPAD + (half << 3)];
  const _Float16* bBase = &sWt[(ct * 16 + rc) * KPAD + (half << 3)];

  float8 acc = {0.f, 0.f, 0.f, 0.f, 0.f, 0.f, 0.f, 0.f};
#pragma unroll
  for (int ks = 0; ks < KPAD / 32; ++ks) {
    // A frag: lane holds row (rt*16+rc); elements 0..7 = K base..base+7,
    // elements 8..15 = K base+16..base+23, base = 8*half + 32*ks.
    half8 a0 = *(const half8*)(aBase + ks * 32);
    half8 a1 = *(const half8*)(aBase + ks * 32 + 16);
    // B frag: lane holds col (ct*16+rc); elements 0..7 = K base..base+7,
    // elements 8..15 = K base+16..base+23, base = 8*half + 32*ks.
    half8 b0 = *(const half8*)(bBase + ks * 32);
    half8 b1 = *(const half8*)(bBase + ks * 32 + 16);
    half16 a, b;
#pragma unroll
    for (int e = 0; e < 8; ++e) {
      a[e] = a0[e]; a[e + 8] = a1[e];
      b[e] = b0[e]; b[e + 8] = b1[e];
    }
    acc = __builtin_amdgcn_wmma_f32_16x16x32_f16(
        /*neg_a=*/false, a, /*neg_b=*/false, b,
        /*c_mod=*/(short)0, acc, /*reuse_a=*/false, /*reuse_b=*/false);
  }

  // C/D layout: lanes 0-15 -> M=r, lanes 16-31 -> M=r+8; N = lane%16
  const int col = ct * 16 + rc;
  float* o = Out + (size_t)(row0 + rt * 16 + (half << 3)) * NC + col;
  if constexpr (HAS_BIAS) {
    const float bv = bias[col];
#pragma unroll
    for (int r = 0; r < 8; ++r) o[(size_t)r * NC] = acc[r] + bv;
  } else {
#pragma unroll
    for (int r = 0; r < 8; ++r) o[(size_t)r * NC] = acc[r];
  }
}

// ---------------------------------------------------------------------------
// conv1 epilogue (C=16): a = (self + 20 gathered neighbors)/21 + bias;
// out = relu(bn(a)).  Thread per (row, channel).
// ---------------------------------------------------------------------------
__global__ __launch_bounds__(256)
void aggr16_bn_relu(const float* __restrict__ h, const int* __restrict__ idx,
                    const float* __restrict__ bias, const float* __restrict__ g,
                    const float* __restrict__ be, float* __restrict__ out) {
  int t = blockIdx.x * blockDim.x + threadIdx.x;   // grid sized to M*16 exactly
  int i = t >> 4, c = t & 15;
  int n = i % NN;
  int base = i - n;                    // batch offset (b*NN)
  float s = h[(size_t)i * 16 + c];
  const int* nb = idx + n * TOPK;
#pragma unroll
  for (int k = 0; k < TOPK; ++k) s += h[(size_t)(base + nb[k]) * 16 + c];
  float a = s * (1.0f / 21.0f) + bias[c];
  float v = a * (g[c] * bn_inv()) + be[c];
  out[t] = fmaxf(v, 0.0f);
}

// ---------------------------------------------------------------------------
// conv2 epilogue (C=64): a = aggregate/21 + b2; z = bn2(a);
// ls = log_softmax(z); y = ls * emb[n]; out = relu(bn_out(y)).
// Block = (64 channels) x (4 rows); LDS tree reductions for max & sum.
// ---------------------------------------------------------------------------
__global__ __launch_bounds__(256)
void aggr2_lsm(const float* __restrict__ h, const int* __restrict__ idx,
               const float* __restrict__ b2, const float* __restrict__ g2,
               const float* __restrict__ be2, const float* __restrict__ emb,
               const float* __restrict__ bg, const float* __restrict__ bb,
               float* __restrict__ out) {
  __shared__ float red[4][64];
  const int c  = threadIdx.x;          // 0..63
  const int ty = threadIdx.y;          // 0..3
  const int row = blockIdx.x * 4 + ty; // M divisible by 4
  int n = row % NN;
  int base = row - n;
  float s = h[(size_t)row * 64 + c];
  const int* nb = idx + n * TOPK;
#pragma unroll
  for (int k = 0; k < TOPK; ++k) s += h[(size_t)(base + nb[k]) * 64 + c];
  float a = s * (1.0f / 21.0f) + b2[c];
  float z = a * (g2[c] * bn_inv()) + be2[c];

  red[ty][c] = z;
  __syncthreads();
  for (int off = 32; off > 0; off >>= 1) {
    if (c < off) red[ty][c] = fmaxf(red[ty][c], red[ty][c + off]);
    __syncthreads();
  }
  float m = red[ty][0];
  __syncthreads();
  red[ty][c] = expf(z - m);
  __syncthreads();
  for (int off = 32; off > 0; off >>= 1) {
    if (c < off) red[ty][c] += red[ty][c + off];
    __syncthreads();
  }
  float l = logf(red[ty][0]);

  float ls = z - m - l;
  float y  = ls * emb[n * 64 + c];
  float o  = y * (bg[c] * bn_inv()) + bb[c];
  out[(size_t)row * 64 + c] = fmaxf(o, 0.0f);
}

// ---------------------------------------------------------------------------
// Tiny output heads: out[i][j] = x[i][:64] . W[:,j] + b[j], O in {1,3}.
// ---------------------------------------------------------------------------
template <int O>
__global__ __launch_bounds__(256)
void matvec_kernel(const float* __restrict__ x, const float* __restrict__ W,
                   const float* __restrict__ b, float* __restrict__ out, int M) {
  int i = blockIdx.x * blockDim.x + threadIdx.x;
  if (i >= M) return;
  float acc[O];
#pragma unroll
  for (int j = 0; j < O; ++j) acc[j] = 0.0f;
  for (int c = 0; c < 64; ++c) {
    float xv = x[(size_t)i * 64 + c];
#pragma unroll
    for (int j = 0; j < O; ++j) acc[j] += xv * W[c * O + j];
  }
#pragma unroll
  for (int j = 0; j < O; ++j) out[(size_t)i * O + j] = acc[j] + b[j];
}

// ---------------------------------------------------------------------------
extern "C" void kernel_launch(void* const* d_in, const int* in_sizes, int n_in,
                              void* d_out, int out_size, void* d_ws, size_t ws_size,
                              hipStream_t stream) {
  (void)in_sizes; (void)n_in; (void)out_size; (void)ws_size;
  const float* data    = (const float*)d_in[0];
  // d_in[1..3] = phy/net/mul edge_index: dead args in reference forward
  const float* mul_emb = (const float*)d_in[4];
  const float* phy_emb = (const float*)d_in[5];
  const float* net_emb = (const float*)d_in[6];
  const float* sW1 = (const float*)d_in[7],  *sb1 = (const float*)d_in[8];
  const float* sg1 = (const float*)d_in[9],  *sbe1 = (const float*)d_in[10];
  const float* sW2 = (const float*)d_in[11], *sb2 = (const float*)d_in[12];
  const float* sg2 = (const float*)d_in[13], *sbe2 = (const float*)d_in[14];
  const float* pW1 = (const float*)d_in[15], *pb1 = (const float*)d_in[16];
  const float* pg1 = (const float*)d_in[17], *pbe1 = (const float*)d_in[18];
  const float* pW2 = (const float*)d_in[19], *pb2 = (const float*)d_in[20];
  const float* pg2 = (const float*)d_in[21], *pbe2 = (const float*)d_in[22];
  const float* nW1 = (const float*)d_in[23], *nb1 = (const float*)d_in[24];
  const float* ng1 = (const float*)d_in[25], *nbe1 = (const float*)d_in[26];
  const float* nW2 = (const float*)d_in[27], *nb2 = (const float*)d_in[28];
  const float* ng2 = (const float*)d_in[29], *nbe2 = (const float*)d_in[30];
  const float* bn_g = (const float*)d_in[31], *bn_b = (const float*)d_in[32];
  const float* pbn_g = (const float*)d_in[33], *pbn_b = (const float*)d_in[34];
  const float* nbn_g = (const float*)d_in[35], *nbn_b = (const float*)d_in[36];
  const float* outW = (const float*)d_in[37], *outb = (const float*)d_in[38];
  const float* poutW = (const float*)d_in[39], *poutb = (const float*)d_in[40];
  const float* noutW = (const float*)d_in[41], *noutb = (const float*)d_in[42];

  const int M = MM;
  float* ws = (float*)d_ws;
  float* A  = ws;                         // [M x 64]
  float* Bb = A  + (size_t)64 * M;        // [M x 64]
  float* Cc = Bb + (size_t)64 * M;        // [M x 64]  (out2, kept live)
  float* D  = Cc + (size_t)64 * M;        // [M x 16]
  float* E  = D  + (size_t)16 * M;        // [M x 16]
  int* idxm = (int*)(E + (size_t)16 * M); // [N x 20] each
  int* idxp = idxm + NN * TOPK;
  int* idxn = idxp + NN * TOPK;

  float* out_phy = (float*)d_out;         // [M x 1]
  float* out_net = out_phy + M;           // [M x 3]

  dim3 b256(256), b64x4(64, 4);

  // learned graphs
  topk_kernel<<<NN, b256, 0, stream>>>(mul_emb, idxm);
  topk_kernel<<<NN, b256, 0, stream>>>(phy_emb, idxp);
  topk_kernel<<<NN, b256, 0, stream>>>(net_emb, idxn);

  // ---- share branch: 60->16 -> aggr -> 16->64 -> aggr2 -> @outW ----
  wmma_gemm<64, 16, 128, false><<<M / 128, b256, 0, stream>>>(data, 60, 0, 60, sW1, nullptr, D);
  aggr16_bn_relu<<<(M * 16) / 256, b256, 0, stream>>>(D, idxm, sb1, sg1, sbe1, E);
  wmma_gemm<32, 64, 32, false><<<M / 32, b256, 0, stream>>>(E, 16, 0, 16, sW2, nullptr, A);
  aggr2_lsm<<<M / 4, b64x4, 0, stream>>>(A, idxm, sb2, sg2, sbe2, mul_emb, bn_g, bn_b, Bb);
  wmma_gemm<64, 64, 32, true><<<M / 32, b256, 0, stream>>>(Bb, 64, 0, 64, outW, outb, Cc);

  // ---- phy branch: out2[:, :16] ----
  wmma_gemm<32, 16, 128, false><<<M / 128, b256, 0, stream>>>(Cc, 64, 0, 16, pW1, nullptr, D);
  aggr16_bn_relu<<<(M * 16) / 256, b256, 0, stream>>>(D, idxp, pb1, pg1, pbe1, E);
  wmma_gemm<32, 64, 32, false><<<M / 32, b256, 0, stream>>>(E, 16, 0, 16, pW2, nullptr, A);
  aggr2_lsm<<<M / 4, b64x4, 0, stream>>>(A, idxp, pb2, pg2, pbe2, phy_emb, pbn_g, pbn_b, Bb);
  matvec_kernel<1><<<(M + 255) / 256, b256, 0, stream>>>(Bb, poutW, poutb, out_phy, M);

  // ---- net branch: out2[:, 16:64] ----
  wmma_gemm<64, 16, 128, false><<<M / 128, b256, 0, stream>>>(Cc, 64, 16, 48, nW1, nullptr, D);
  aggr16_bn_relu<<<(M * 16) / 256, b256, 0, stream>>>(D, idxn, nb1, ng1, nbe1, E);
  wmma_gemm<32, 64, 32, false><<<M / 32, b256, 0, stream>>>(E, 16, 0, 16, nW2, nullptr, A);
  aggr2_lsm<<<M / 4, b64x4, 0, stream>>>(A, idxn, nb2, ng2, nbe2, net_emb, nbn_g, nbn_b, Bb);
  matvec_kernel<3><<<(M + 255) / 256, b256, 0, stream>>>(Bb, noutW, noutb, out_net, M);
}
